// MoE_7043746365556
// MI455X (gfx1250) — compile-verified, matrix operand order checked by reading, not run on hardware
//
#include <hip/hip_runtime.h>
#include <hip/hip_bf16.h>

// ---------------------------------------------------------------------------
// MoE (8 experts, top-2) grouped-GEMM implementation for gfx1250 (MI455X).
//   gate -> route lists -> bf16 weight transpose -> 3x WMMA GEMM passes
//   -> combine (gated sum of the two expert outputs per token).
// Matrix core path: v_wmma_f32_16x16x32_bf16 (wave32, 16x16x32 tiles).
// Each wave computes a 32x64 output tile: 2 A-frags x 4 B-frags -> 8 WMMA
// per k-step (B frags reused from registers across both M-frags).
// ---------------------------------------------------------------------------

#define EMB    768
#define HID    3072
#define HID2   6144
#define NE     8
#define TOKENS 16384            // B*N = 8*2048

typedef __attribute__((ext_vector_type(16))) __bf16       v16bf;
typedef __attribute__((ext_vector_type(8)))  float        v8f;
typedef __attribute__((ext_vector_type(4)))  unsigned int u32x4;

union Frag { u32x4 u[2]; v16bf v; };   // 32 bytes = one 16x32-bf16 A/B fragment per lane

__device__ __forceinline__ float gelu_erf(float v) {
    return 0.5f * v * (1.0f + erff(v * 0.70710678118654752f));
}

// ---------------------------------------------------------------------------
// init: zero the expert counters (workspace is poisoned by the harness).
// ---------------------------------------------------------------------------
__global__ void init_kernel(int* __restrict__ counts) {
    if (threadIdx.x < NE && blockIdx.x == 0) counts[threadIdx.x] = 0;
}

// ---------------------------------------------------------------------------
// cvt_x: activations f32 -> bf16
// ---------------------------------------------------------------------------
__global__ __launch_bounds__(256) void cvt_x_kernel(const float* __restrict__ x,
                                                    __bf16* __restrict__ xb, int n) {
    const int i = blockIdx.x * 256 + threadIdx.x;
    if (i < n) xb[i] = (__bf16)x[i];
}

// ---------------------------------------------------------------------------
// transpose_cvt: W [e][Kd][Nd] f32  ->  Wt [e][Nd][Kd] bf16  (LDS 32x33 tile)
// N-major storage makes the WMMA B operand a contiguous per-lane read.
// ---------------------------------------------------------------------------
__global__ __launch_bounds__(256) void transpose_cvt_kernel(const float* __restrict__ in,
                                                            __bf16* __restrict__ outp,
                                                            int Kd, int Nd) {
    __shared__ float tile[32][33];
    const int e  = blockIdx.z;
    const int k0 = blockIdx.x * 32;
    const int n0 = blockIdx.y * 32;
    const int tx = threadIdx.x & 31;
    const int ty = threadIdx.x >> 5;          // 8 rows per pass
    const float* src = in + (size_t)e * Kd * Nd;
    __bf16*      dst = outp + (size_t)e * Nd * Kd;
    #pragma unroll
    for (int r = 0; r < 32; r += 8)
        tile[ty + r][tx] = src[(size_t)(k0 + ty + r) * Nd + (n0 + tx)];
    __syncthreads();
    #pragma unroll
    for (int r = 0; r < 32; r += 8)
        dst[(size_t)(n0 + ty + r) * Kd + (k0 + tx)] = (__bf16)tile[tx][ty + r];
}

// ---------------------------------------------------------------------------
// gate: per-token scores = x @ Wg + bg, top-2, softmax over the 2 logits.
// Scatters token ids into per-expert lists (for the A-gather of GEMM1)
// and records per-token (expert, slot, gate) for the final combine.
// ---------------------------------------------------------------------------
__global__ __launch_bounds__(256) void gate_kernel(const float* __restrict__ x,
                                                   const float* __restrict__ Wg,
                                                   const float* __restrict__ bg,
                                                   int* __restrict__ counts,
                                                   int* __restrict__ route_tok,
                                                   int* __restrict__ tok_e,
                                                   int* __restrict__ tok_i,
                                                   float* __restrict__ tok_g) {
    __shared__ float sWg[EMB * NE];           // 24 KB
    for (int i = threadIdx.x; i < EMB * NE; i += 256) sWg[i] = Wg[i];
    __syncthreads();

    const int t = blockIdx.x * 256 + threadIdx.x;   // TOKENS % 256 == 0
    const float* xr = x + (size_t)t * EMB;
    float acc[NE];
    #pragma unroll
    for (int e = 0; e < NE; ++e) acc[e] = bg[e];
    for (int k = 0; k < EMB; ++k) {
        const float xv = xr[k];
        #pragma unroll
        for (int e = 0; e < NE; ++e) acc[e] += xv * sWg[k * NE + e];
    }
    // top-2 (ties -> lower index, matching lax.top_k)
    int e0 = 0; float v0 = acc[0];
    #pragma unroll
    for (int e = 1; e < NE; ++e) if (acc[e] > v0) { v0 = acc[e]; e0 = e; }
    int e1 = (e0 == 0) ? 1 : 0; float v1 = acc[e1];
    #pragma unroll
    for (int e = 0; e < NE; ++e)
        if (e != e0 && acc[e] > v1) { v1 = acc[e]; e1 = e; }
    // softmax over {v0, v1}, v0 >= v1
    const float w  = expf(v1 - v0);
    const float g0 = 1.0f / (1.0f + w);
    const float g1 = w / (1.0f + w);

    const int i0 = atomicAdd(counts + e0, 1);
    route_tok[e0 * TOKENS + i0] = t;
    tok_e[2 * t]     = e0;  tok_i[2 * t]     = i0;  tok_g[2 * t]     = g0;
    const int i1 = atomicAdd(counts + e1, 1);
    route_tok[e1 * TOKENS + i1] = t;
    tok_e[2 * t + 1] = e1;  tok_i[2 * t + 1] = i1;  tok_g[2 * t + 1] = g1;
}

// ---------------------------------------------------------------------------
// offsets: exclusive prefix sum over 8 counters (activation pool segments).
// ---------------------------------------------------------------------------
__global__ void offsets_kernel(const int* __restrict__ counts, int* __restrict__ offs) {
    if (threadIdx.x == 0 && blockIdx.x == 0) {
        int s = 0;
        #pragma unroll
        for (int e = 0; e < NE; ++e) { offs[e] = s; s += counts[e]; }
    }
}

// ---------------------------------------------------------------------------
// ffn: one expert-segment GEMM pass.
//   Block = 256 threads = 8 waves arranged 2(M) x 4(N); wave tile = 32x64.
//   Per k-step: 2 A-frag + 4 B-frag b128 loads feed 8 v_wmma_f32_16x16x32_bf16
//   (~21 FLOP per loaded byte at register level).
//   MODE 0: D = gelu(A@Wt^T + bias) stored bf16 to activation pool.
//   MODE 1: D = A@Wt^T + bias stored f32 to y-pool (combined later).
// ---------------------------------------------------------------------------
template <int KDIM, int NDIM, int MODE>
__global__ __launch_bounds__(256) void ffn_kernel(const __bf16* __restrict__ Abase,
                                                  const __bf16* __restrict__ Wt,
                                                  const float*  __restrict__ bias,
                                                  __bf16*       __restrict__ Hout,
                                                  float*        __restrict__ Ypool,
                                                  const int*    __restrict__ counts,
                                                  const int*    __restrict__ offsets,
                                                  const int*    __restrict__ route_tok,
                                                  int gatherA) {
    const int e      = blockIdx.z;
    const int cnt    = counts[e];
    const int m_tile = blockIdx.y * 64;
    if (m_tile >= cnt) return;                    // over-provisioned grid: early exit
    const int base = offsets[e];

    const int lane = threadIdx.x & 31;
    const int wave = threadIdx.x >> 5;
    const int wx   = wave & 3;                    // n position within block
    const int wy   = wave >> 2;                   // m position within block

    const int m0 = m_tile + wy * 32;
    if (m0 >= cnt) return;                        // wave-uniform: EXEC stays all-ones
    const int n0  = blockIdx.x * 256 + wx * 64;
    const int l15 = lane & 15;
    const int c0  = (lane & 16) ? 8 : 0;          // K half selection per ISA layout

    int mr0 = m0 + l15;       if (mr0 >= cnt) mr0 = cnt - 1;
    int mr1 = m0 + 16 + l15;  if (mr1 >= cnt) mr1 = cnt - 1;
    const __bf16 *arow0, *arow1;
    if (gatherA) {
        arow0 = Abase + (size_t)route_tok[e * TOKENS + mr0] * KDIM;
        arow1 = Abase + (size_t)route_tok[e * TOKENS + mr1] * KDIM;
    } else {
        arow0 = Abase + (size_t)(base + mr0) * KDIM;
        arow1 = Abase + (size_t)(base + mr1) * KDIM;
    }
    const __bf16* wrow = Wt + ((size_t)e * NDIM + (n0 + l15)) * KDIM;

    v8f acc[2][4] = {};
    #pragma unroll 2
    for (int kb = 0; kb < KDIM; kb += 32) {
        __builtin_prefetch(wrow + kb + 512, 0, 1);   // -> global_prefetch_b8
        Frag a0, a1;
        a0.u[0] = *(const u32x4*)(arow0 + kb + c0);
        a0.u[1] = *(const u32x4*)(arow0 + kb + c0 + 16);
        a1.u[0] = *(const u32x4*)(arow1 + kb + c0);
        a1.u[1] = *(const u32x4*)(arow1 + kb + c0 + 16);
        #pragma unroll
        for (int j = 0; j < 4; ++j) {
            const __bf16* wj = wrow + (size_t)j * 16 * KDIM;   // folds to imm offset
            Frag b;
            b.u[0] = *(const u32x4*)(wj + kb + c0);
            b.u[1] = *(const u32x4*)(wj + kb + c0 + 16);
            acc[0][j] = __builtin_amdgcn_wmma_f32_16x16x32_bf16(false, a0.v, false, b.v,
                                                                (short)0, acc[0][j], false, false);
            acc[1][j] = __builtin_amdgcn_wmma_f32_16x16x32_bf16(false, a1.v, false, b.v,
                                                                (short)0, acc[1][j], false, false);
        }
    }

    // C/D layout: VGPR i -> M = i + (lane>=16 ? 8 : 0); N = lane & 15.
    const int mhi = (lane & 16) ? 8 : 0;
    #pragma unroll
    for (int h = 0; h < 2; ++h) {
        #pragma unroll
        for (int j = 0; j < 4; ++j) {
            const int   nn = n0 + j * 16 + l15;
            const float bn = bias[e * NDIM + nn];
            #pragma unroll
            for (int i = 0; i < 8; ++i) {
                const int m = m0 + h * 16 + mhi + i;
                if (m < cnt) {
                    const float v = acc[h][j][i] + bn;
                    if (MODE == 0)
                        Hout[(size_t)(base + m) * NDIM + nn] = (__bf16)gelu_erf(v);
                    else
                        Ypool[(size_t)(base + m) * NDIM + nn] = v;
                }
            }
        }
    }
}

// ---------------------------------------------------------------------------
// combine: out[t] = g0 * y[slot(t,0)] + g1 * y[slot(t,1)]  (one block per token)
// ---------------------------------------------------------------------------
__global__ __launch_bounds__(256) void combine_kernel(const float* __restrict__ ypool,
                                                      const int*   __restrict__ offs,
                                                      const int*   __restrict__ tok_e,
                                                      const int*   __restrict__ tok_i,
                                                      const float* __restrict__ tok_g,
                                                      float*       __restrict__ out) {
    const int t  = blockIdx.x;
    const int s0 = offs[tok_e[2 * t]]     + tok_i[2 * t];
    const int s1 = offs[tok_e[2 * t + 1]] + tok_i[2 * t + 1];
    const float g0 = tok_g[2 * t];
    const float g1 = tok_g[2 * t + 1];
    const float* y0 = ypool + (size_t)s0 * EMB;
    const float* y1 = ypool + (size_t)s1 * EMB;
    float* o = out + (size_t)t * EMB;
    for (int n = threadIdx.x; n < EMB; n += 256)
        o[n] = g0 * y0[n] + g1 * y1[n];
}

// ---------------------------------------------------------------------------
// host launcher
// ---------------------------------------------------------------------------
extern "C" void kernel_launch(void* const* d_in, const int* in_sizes, int n_in,
                              void* d_out, int out_size, void* d_ws, size_t ws_size,
                              hipStream_t stream) {
    (void)in_sizes; (void)n_in; (void)ws_size; (void)out_size;
    const float* x  = (const float*)d_in[0];
    const float* Wg = (const float*)d_in[1];
    const float* bg = (const float*)d_in[2];
    const float* W1 = (const float*)d_in[3];
    const float* b1 = (const float*)d_in[4];
    const float* W2 = (const float*)d_in[5];
    const float* b2 = (const float*)d_in[6];
    const float* W3 = (const float*)d_in[7];
    const float* b3 = (const float*)d_in[8];
    float* out = (float*)d_out;

    // ---- workspace layout (256B aligned); total ~= 1.15 GB -----------------
    char* wsp = (char*)d_ws;
    size_t off = 0;
    auto take = [&](size_t bytes) -> void* {
        off = (off + 255) & ~(size_t)255;
        void* p = wsp + off;
        off += bytes;
        return p;
    };
    int*    counts    = (int*)   take(NE * sizeof(int));
    int*    offs      = (int*)   take(NE * sizeof(int));
    int*    route_tok = (int*)   take((size_t)NE * TOKENS * sizeof(int));
    int*    tok_e     = (int*)   take((size_t)2 * TOKENS * sizeof(int));
    int*    tok_i     = (int*)   take((size_t)2 * TOKENS * sizeof(int));
    float*  tok_g     = (float*) take((size_t)2 * TOKENS * sizeof(float));
    __bf16* xb        = (__bf16*)take((size_t)TOKENS * EMB * 2);
    __bf16* w1t       = (__bf16*)take((size_t)NE * HID  * EMB  * 2);
    __bf16* w2t       = (__bf16*)take((size_t)NE * HID2 * HID  * 2);
    __bf16* w3t      = (__bf16*)take((size_t)NE * EMB  * HID2 * 2);
    __bf16* h1        = (__bf16*)take((size_t)2 * TOKENS * HID  * 2);
    __bf16* h2        = (__bf16*)take((size_t)2 * TOKENS * HID2 * 2);
    float*  ypool     = (float*) take((size_t)2 * TOKENS * EMB * 4);

    // 1) zero expert counters
    init_kernel<<<dim3(1), dim3(32), 0, stream>>>(counts);

    // 2) x -> bf16
    cvt_x_kernel<<<dim3((TOKENS * EMB) / 256), dim3(256), 0, stream>>>(x, xb, TOKENS * EMB);

    // 3) weights -> bf16, N-major [e][n][k]
    transpose_cvt_kernel<<<dim3(EMB / 32,  HID  / 32, NE), dim3(256), 0, stream>>>(W1, w1t, EMB,  HID);
    transpose_cvt_kernel<<<dim3(HID / 32,  HID2 / 32, NE), dim3(256), 0, stream>>>(W2, w2t, HID,  HID2);
    transpose_cvt_kernel<<<dim3(HID2 / 32, EMB  / 32, NE), dim3(256), 0, stream>>>(W3, w3t, HID2, EMB);

    // 4) gating + routing
    gate_kernel<<<dim3(TOKENS / 256), dim3(256), 0, stream>>>(x, Wg, bg, counts,
                                                              route_tok, tok_e, tok_i, tok_g);
    // 5) pool segment offsets
    offsets_kernel<<<dim3(1), dim3(32), 0, stream>>>(counts, offs);

    // 6) three WMMA GEMM passes (grid over-provisioned for worst-case routing)
    const int mT = (TOKENS + 63) / 64;               // 256 m-tiles max per expert
    ffn_kernel<EMB, HID, 0><<<dim3(HID / 256, mT, NE), dim3(256), 0, stream>>>(
        xb, w1t, b1, h1, nullptr, counts, offs, route_tok, 1);
    ffn_kernel<HID, HID2, 0><<<dim3(HID2 / 256, mT, NE), dim3(256), 0, stream>>>(
        h1, w2t, b2, h2, nullptr, counts, offs, route_tok, 0);
    ffn_kernel<HID2, EMB, 1><<<dim3(EMB / 256, mT, NE), dim3(256), 0, stream>>>(
        h2, w3t, b3, nullptr, ypool, counts, offs, route_tok, 0);

    // 7) gated combine of the two expert outputs per token
    combine_kernel<<<dim3(TOKENS), dim3(256), 0, stream>>>(ypool, offs, tok_e, tok_i,
                                                           tok_g, out);
}
